// GAT_39453569581650
// MI455X (gfx1250) — compile-verified, matrix operand order checked by reading, not run on hardware
//
#include <hip/hip_runtime.h>

// ---------------- problem constants (match reference) ----------------
#define Bn   8
#define Nn   5000
#define En   80000
#define KIN  30
#define Hh   4
#define Dd   64
#define HDc  256
#define OUTc 3
#define Mr   (Bn * Nn)   // 40000 rows
#define EPSc 1e-5f

typedef __attribute__((ext_vector_type(16))) __bf16        v16bf;
typedef __attribute__((ext_vector_type(8)))  float         v8f;
typedef __attribute__((ext_vector_type(4)))  unsigned int  u32x4;

union Frag { v16bf v; u32x4 q[2]; };

__device__ __forceinline__ unsigned short f2bf(float f) {
  unsigned u = __float_as_uint(f);
  u += 0x7FFFu + ((u >> 16) & 1u);          // round-to-nearest-even
  return (unsigned short)(u >> 16);
}
__device__ __forceinline__ float lrelu(float x, float a) { return x > 0.f ? x : a * x; }

__device__ __forceinline__ void atomicMaxF(float* addr, float v) {
  if (v >= 0.f) atomicMax((int*)addr, __float_as_int(v));
  else          atomicMin((unsigned int*)addr, (unsigned int)__float_as_int(v));
}

// ---------------- utility kernels ----------------
__global__ void k_copy_f32(const float* __restrict__ s, float* __restrict__ d, int n) {
  int i = blockIdx.x * blockDim.x + threadIdx.x;
  if (i < n) d[i] = s[i];
}
__global__ void k_zero_f32(float* __restrict__ d, int n) {
  int i = blockIdx.x * blockDim.x + threadIdx.x;
  if (i < n) d[i] = 0.f;
}
__global__ void k_init_ms(float* __restrict__ m, float* __restrict__ s, int n) {
  int i = blockIdx.x * blockDim.x + threadIdx.x;
  if (i < n) { m[i] = __int_as_float(0xFF800000); s[i] = 0.f; }  // -inf, 0
}

// W [K,HD] f32  ->  Wt [HD,Kp] bf16 (column-major of W, zero-padded K)
__global__ void k_prep_w(const float* __restrict__ W, unsigned short* __restrict__ Wt,
                         int K, int Kp) {
  int i = blockIdx.x * blockDim.x + threadIdx.x;
  if (i >= HDc * Kp) return;
  int n = i / Kp, k = i % Kp;
  float v = (k < K) ? W[k * HDc + n] : 0.f;
  Wt[n * Kp + k] = f2bf(v);
}

// X [M,K] f32 (dense rows, stride K) -> Xb [M,Kp] bf16, zero-padded
__global__ void k_conv_x(const float* __restrict__ X, unsigned short* __restrict__ Xb,
                         int K, int Kp) {
  long i = (long)blockIdx.x * blockDim.x + threadIdx.x;
  if (i >= (long)Mr * Kp) return;
  int m = (int)(i / Kp), k = (int)(i % Kp);
  float v = (k < K) ? X[(long)m * K + k] : 0.f;
  Xb[(long)m * Kp + k] = f2bf(v);
}

// ---------------- WMMA GEMM (layer 0, K=32): direct-global fragments ----------------
// One wave per 16x64 output strip; A fragment reused across 4 WMMAs.
template <int KP>
__global__ __launch_bounds__(32) void k_wmma_gemm(
    const unsigned short* __restrict__ Xb,
    const unsigned short* __restrict__ Wt,
    float* __restrict__ out) {
  const int lane = threadIdx.x & 31;
  const int half = lane >> 4;          // 0 or 1
  const int lidx = lane & 15;
  const int mt   = blockIdx.x;         // M/16 tiles
  const int nt0  = blockIdx.y << 2;    // 4 N-tiles per wave

  const unsigned short* arow = Xb + (size_t)(mt * 16 + lidx) * KP;
  v8f z = {0.f, 0.f, 0.f, 0.f, 0.f, 0.f, 0.f, 0.f};
  v8f acc[4] = {z, z, z, z};

#pragma unroll
  for (int kt = 0; kt < KP; kt += 32) {
    // A 16x32 bf16 fragment: lanes 0-15 hold K {0..7,16..23}, lanes 16-31 {8..15,24..31}
    Frag a;
    a.q[0] = *(const u32x4*)(arow + kt + half * 8);
    a.q[1] = *(const u32x4*)(arow + kt + 16 + half * 8);
    if (kt + 32 < KP) __builtin_prefetch(arow + kt + 32, 0, 3);  // global_prefetch
#pragma unroll
    for (int j = 0; j < 4; ++j) {
      const unsigned short* bcol = Wt + (size_t)((nt0 + j) * 16 + lidx) * KP;
      Frag b;
      b.q[0] = *(const u32x4*)(bcol + kt + half * 8);
      b.q[1] = *(const u32x4*)(bcol + kt + 16 + half * 8);
      acc[j] = __builtin_amdgcn_wmma_f32_16x16x32_bf16(
          false, a.v, false, b.v, (short)0, acc[j], false, false);
    }
  }
#pragma unroll
  for (int j = 0; j < 4; ++j) {
    float* od = out + (size_t)(mt * 16 + half * 8) * HDc + (nt0 + j) * 16 + lidx;
#pragma unroll
    for (int v = 0; v < 8; ++v) od[(size_t)v * HDc] = acc[j][v];
  }
}

// ---------------- WMMA GEMM (layers 1-3, K=256): async B-stage into LDS --------------
// Block = 4 waves. The block's 64-column x 256-K bf16 weight slice (32 KB) is copied
// into LDS once with global_load_async_to_lds_b128 (ASYNCcnt), then each wave computes
// one 16x64 output strip, pulling B fragments from LDS (ds_load_b128) and streaming A
// fragments from global.
__global__ __launch_bounds__(128) void k_wmma_gemm_lds(
    const unsigned short* __restrict__ Xb,
    const unsigned short* __restrict__ Wt,
    float* __restrict__ out) {
  __shared__ __align__(16) unsigned short ldsB[64 * 256];  // 32 KB
  const int tid  = threadIdx.x;
  const int wave = tid >> 5;
  const int lane = tid & 31;
  const int half = lane >> 4;
  const int lidx = lane & 15;
  const int colBase = blockIdx.y * 64;

  // ---- async stage of B slice: 2048 x 16B chunks, 16 per thread ----
#pragma unroll
  for (int c = tid; c < 64 * 256 / 8; c += 128) {
    int col  = c >> 5;             // 32 chunks per column
    int koff = (c & 31) << 3;      // chunk offset in shorts
    unsigned lo =
        (unsigned)(size_t)(const void*)(ldsB + col * 256 + koff);  // low 32b = LDS offset
    unsigned long long ga =
        (unsigned long long)(const void*)(Wt + (size_t)(colBase + col) * 256 + koff);
    asm volatile("global_load_async_to_lds_b128 %0, %1, off"
                 :: "v"(lo), "v"(ga) : "memory");
  }
  asm volatile("s_wait_asynccnt 0x0" ::: "memory");
  __syncthreads();

  const int mt = blockIdx.x * 4 + wave;
  const unsigned short* arow = Xb + (size_t)(mt * 16 + lidx) * 256;
  v8f z = {0.f, 0.f, 0.f, 0.f, 0.f, 0.f, 0.f, 0.f};
  v8f acc[4] = {z, z, z, z};

#pragma unroll
  for (int kt = 0; kt < 256; kt += 32) {
    Frag a;
    a.q[0] = *(const u32x4*)(arow + kt + half * 8);
    a.q[1] = *(const u32x4*)(arow + kt + 16 + half * 8);
    if (kt + 32 < 256) __builtin_prefetch(arow + kt + 32, 0, 3);
#pragma unroll
    for (int j = 0; j < 4; ++j) {
      const unsigned short* bcol = ldsB + (j * 16 + lidx) * 256;   // LDS-resident column
      Frag b;
      b.q[0] = *(const u32x4*)(bcol + kt + half * 8);
      b.q[1] = *(const u32x4*)(bcol + kt + 16 + half * 8);
      acc[j] = __builtin_amdgcn_wmma_f32_16x16x32_bf16(
          false, a.v, false, b.v, (short)0, acc[j], false, false);
    }
  }
#pragma unroll
  for (int j = 0; j < 4; ++j) {
    float* od = out + (size_t)(mt * 16 + half * 8) * HDc + colBase + j * 16 + lidx;
#pragma unroll
    for (int v = 0; v < 8; ++v) od[(size_t)v * HDc] = acc[j][v];
  }
}

// ---------------- attention ----------------
__global__ void k_attn_logits(const float* __restrict__ hf,
                              const float* __restrict__ al, const float* __restrict__ ar,
                              float* __restrict__ el, float* __restrict__ er) {
  int i = blockIdx.x * blockDim.x + threadIdx.x;
  if (i >= Mr * Hh) return;
  int m = i / Hh, h = i % Hh;
  const float* hp = hf + (size_t)m * HDc + h * Dd;
  const float* alp = al + h * Dd;
  const float* arp = ar + h * Dd;
  float sl = 0.f, sr = 0.f;
#pragma unroll 8
  for (int d = 0; d < Dd; ++d) { float v = hp[d]; sl += v * alp[d]; sr += v * arp[d]; }
  el[i] = sl; er[i] = sr;
}

__global__ void k_edge_logits(const float* __restrict__ el, const float* __restrict__ er,
                              const int* __restrict__ src, const int* __restrict__ dst,
                              float* __restrict__ ebuf, float* __restrict__ mbuf) {
  int t = blockIdx.x * blockDim.x + threadIdx.x;
  if (t >= Bn * En * Hh) return;
  int b = t / (En * Hh); int r = t % (En * Hh);
  int e = r / Hh, h = r % Hh;
  int s = src[e], d = dst[e];
  float v = lrelu(el[(b * Nn + s) * Hh + h] + er[(b * Nn + d) * Hh + h], 0.2f);
  ebuf[t] = v;
  atomicMaxF(&mbuf[(b * Nn + d) * Hh + h], v);
}

__global__ void k_edge_exp(float* __restrict__ ebuf, const float* __restrict__ mbuf,
                           float* __restrict__ sbuf, const int* __restrict__ dst) {
  int t = blockIdx.x * blockDim.x + threadIdx.x;
  if (t >= Bn * En * Hh) return;
  int b = t / (En * Hh); int r = t % (En * Hh);
  int e = r / Hh, h = r % Hh;
  int d = dst[e];
  float ex = __expf(ebuf[t] - mbuf[(b * Nn + d) * Hh + h]);
  ebuf[t] = ex;
  atomicAdd(&sbuf[(b * Nn + d) * Hh + h], ex);
}

__global__ __launch_bounds__(256) void k_edge_scatter(
    const float* __restrict__ ebuf, const float* __restrict__ sbuf,
    const float* __restrict__ hf, const int* __restrict__ src,
    const int* __restrict__ dst, float* __restrict__ gout) {
  int be = blockIdx.x;          // 0..B*E-1
  int b = be / En, e = be % En;
  int h = threadIdx.x >> 6, d = threadIdx.x & 63;
  int s = src[e], ds = dst[e];
  float alpha = ebuf[(size_t)be * Hh + h] / sbuf[(b * Nn + ds) * Hh + h];
  float v = alpha * hf[(size_t)(b * Nn + s) * HDc + h * Dd + d];
  atomicAdd(&gout[(size_t)(b * Nn + ds) * HDc + h * Dd + d], v);
}

// ---------------- batch norm (training stats over B*N) ----------------
__global__ __launch_bounds__(256) void k_bn_stats(const float* __restrict__ x,
                                                  float* __restrict__ mu,
                                                  float* __restrict__ var) {
  int c = blockIdx.x;  // channel 0..255
  float s = 0.f, ss = 0.f;
  for (int m = threadIdx.x; m < Mr; m += 256) {
    float v = x[(size_t)m * HDc + c];
    s += v; ss += v * v;
  }
  __shared__ float sh[256], sh2[256];
  sh[threadIdx.x] = s; sh2[threadIdx.x] = ss;
  __syncthreads();
  for (int o = 128; o > 0; o >>= 1) {
    if (threadIdx.x < o) { sh[threadIdx.x] += sh[threadIdx.x + o]; sh2[threadIdx.x] += sh2[threadIdx.x + o]; }
    __syncthreads();
  }
  if (threadIdx.x == 0) {
    float mean = sh[0] / (float)Mr;
    mu[c] = mean;
    var[c] = sh2[0] / (float)Mr - mean * mean;   // biased, matches torch BN
  }
}

__global__ void k_bn_apply(const float* __restrict__ x, const float* __restrict__ mu,
                           const float* __restrict__ var, const float* __restrict__ g,
                           const float* __restrict__ be, float* __restrict__ y) {
  long i = (long)blockIdx.x * blockDim.x + threadIdx.x;
  if (i >= (long)Mr * HDc) return;
  int c = (int)(i % HDc);
  float v = (x[i] - mu[c]) * rsqrtf(var[c] + EPSc) * g[c] + be[c];
  y[i] = lrelu(v, 0.01f);
}

// ---------------- output projection + rollout window shift ----------------
__global__ void k_out_proj(const float* __restrict__ h, const float* __restrict__ Wo,
                           const float* __restrict__ bo, float* __restrict__ ostep,
                           float* __restrict__ dout, int t, int T) {
  int i = blockIdx.x * blockDim.x + threadIdx.x;
  if (i >= Mr * OUTc) return;
  int m = i / OUTc, o = i % OUTc;
  const float* hp = h + (size_t)m * HDc;
  float acc = bo[o];
#pragma unroll 8
  for (int j = 0; j < HDc; ++j) acc += hp[j] * Wo[j * OUTc + o];
  ostep[i] = acc;
  dout[((size_t)m * T + t) * OUTc + o] = acc;   // [B,N,T,OUT]
}

__global__ void k_shift(const float* __restrict__ xold, const float* __restrict__ ostep,
                        float* __restrict__ xnew) {
  long i = (long)blockIdx.x * blockDim.x + threadIdx.x;
  if (i >= (long)Mr * KIN) return;
  int m = (int)(i / KIN), c = (int)(i % KIN);
  xnew[i] = (c < KIN - OUTc) ? xold[(long)m * KIN + c + OUTc]
                             : ostep[m * OUTc + (c - (KIN - OUTc))];
}

// ---------------- host orchestration ----------------
extern "C" void kernel_launch(void* const* d_in, const int* in_sizes, int n_in,
                              void* d_out, int out_size, void* d_ws, size_t ws_size,
                              hipStream_t stream) {
  (void)in_sizes; (void)n_in; (void)ws_size;
  const float* xx  = (const float*)d_in[0];
  const int*   src = (const int*)d_in[1];
  const int*   dst = (const int*)d_in[2];
  const float *W[4], *al[4], *ar[4], *gm[4], *bt[4];
  for (int l = 0; l < 4; ++l) {
    W[l]  = (const float*)d_in[3 + 5 * l];
    al[l] = (const float*)d_in[4 + 5 * l];
    ar[l] = (const float*)d_in[5 + 5 * l];
    gm[l] = (const float*)d_in[6 + 5 * l];
    bt[l] = (const float*)d_in[7 + 5 * l];
  }
  const float* Wout = (const float*)d_in[23];
  const float* bout = (const float*)d_in[24];
  const int T = out_size / (Mr * OUTc);   // output_length (device scalar not readable here)

  const int Kdim[4] = {KIN, HDc, HDc, HDc};
  const int Kp[4]   = {32,  HDc, HDc, HDc};

  // carve workspace
  char* p = (char*)d_ws;
  auto carve = [&](size_t bytes) -> void* {
    void* r = (void*)p; p += (bytes + 255) & ~(size_t)255; return r;
  };
  float* actA  = (float*)carve((size_t)Mr * HDc * 4);   // layer input / BN output
  float* actB  = (float*)carve((size_t)Mr * HDc * 4);   // gemm out (h features)
  float* actC  = (float*)carve((size_t)Mr * HDc * 4);   // attention-aggregated output
  unsigned short* Xb = (unsigned short*)carve((size_t)Mr * HDc * 2);
  unsigned short* Wt[4];
  for (int l = 0; l < 4; ++l) Wt[l] = (unsigned short*)carve((size_t)HDc * Kp[l] * 2);
  float* elb  = (float*)carve((size_t)Mr * Hh * 4);
  float* erb  = (float*)carve((size_t)Mr * Hh * 4);
  float* mbuf = (float*)carve((size_t)Mr * Hh * 4);
  float* sbuf = (float*)carve((size_t)Mr * Hh * 4);
  float* ebuf = (float*)carve((size_t)Bn * En * Hh * 4);
  float* xxA  = (float*)carve((size_t)Mr * KIN * 4);
  float* xxB  = (float*)carve((size_t)Mr * KIN * 4);
  float* ostp = (float*)carve((size_t)Mr * OUTc * 4);
  float* bnmu = (float*)carve(HDc * 4);
  float* bnvr = (float*)carve(HDc * 4);

  const int TB = 256;
  auto blk = [](long n, int tb) { return (int)((n + tb - 1) / tb); };

  // once per call: copy rolling input window, convert weights to bf16 column-major
  k_copy_f32<<<blk((long)Mr * KIN, TB), TB, 0, stream>>>(xx, xxA, Mr * KIN);
  for (int l = 0; l < 4; ++l)
    k_prep_w<<<blk((long)HDc * Kp[l], TB), TB, 0, stream>>>(W[l], Wt[l], Kdim[l], Kp[l]);

  float* xcur = xxA;
  float* xnxt = xxB;
  for (int t = 0; t < T; ++t) {
    for (int l = 0; l < 4; ++l) {
      const float* lin = (l == 0) ? xcur : actA;
      k_conv_x<<<blk((long)Mr * Kp[l], TB), TB, 0, stream>>>(lin, Xb, Kdim[l], Kp[l]);
      if (Kp[l] == 32) {
        dim3 gg(Mr / 16, HDc / 64);
        k_wmma_gemm<32><<<gg, 32, 0, stream>>>(Xb, Wt[l], actB);
      } else {
        dim3 gg(Mr / 16 / 4, HDc / 64);   // 4 waves (m-tiles) per block
        k_wmma_gemm_lds<<<gg, 128, 0, stream>>>(Xb, Wt[l], actB);
      }
      k_attn_logits<<<blk((long)Mr * Hh, TB), TB, 0, stream>>>(actB, al[l], ar[l], elb, erb);
      k_init_ms<<<blk((long)Mr * Hh, TB), TB, 0, stream>>>(mbuf, sbuf, Mr * Hh);
      k_zero_f32<<<blk((long)Mr * HDc, TB), TB, 0, stream>>>(actC, Mr * HDc);
      k_edge_logits<<<blk((long)Bn * En * Hh, TB), TB, 0, stream>>>(elb, erb, src, dst, ebuf, mbuf);
      k_edge_exp<<<blk((long)Bn * En * Hh, TB), TB, 0, stream>>>(ebuf, mbuf, sbuf, dst);
      k_edge_scatter<<<Bn * En, 256, 0, stream>>>(ebuf, sbuf, actB, src, dst, actC);
      k_bn_stats<<<HDc, 256, 0, stream>>>(actC, bnmu, bnvr);
      k_bn_apply<<<blk((long)Mr * HDc, TB), TB, 0, stream>>>(actC, bnmu, bnvr, gm[l], bt[l], actA);
    }
    k_out_proj<<<blk((long)Mr * OUTc, TB), TB, 0, stream>>>(actA, Wout, bout, ostp,
                                                            (float*)d_out, t, T);
    k_shift<<<blk((long)Mr * KIN, TB), TB, 0, stream>>>(xcur, ostp, xnxt);
    float* tmp = xcur; xcur = xnxt; xnxt = tmp;
  }
}
